// GNN_23596550324675
// MI455X (gfx1250) — matrix operand order repaired and checked by reading, not compiled, not run on hardware
//
#include <hip/hip_runtime.h>
#include <math.h>

// ---------------------------------------------------------------------------
// Problem constants (match the reference)
// ---------------------------------------------------------------------------
#define NN 100000
#define EE 200000
#define GG 4096
#define DD 300
#define LL 5
#define HH 600      // 2*D
#define DHH 1200    // 4*D
#define MAXO 1783
#define BN_EPS 1e-5f

typedef __attribute__((ext_vector_type(16))) __bf16 v16bf;
typedef __attribute__((ext_vector_type(8)))  __bf16 v8bf;
typedef __attribute__((ext_vector_type(8)))  float  v8f;

// A-side element transforms fused into the GEMM A-tile staging
#define AMODE_NONE    0
#define AMODE_BN_RELU 1
#define AMODE_BN_GELU 2
#define AMODE_SILU    3

// ---------------------------------------------------------------------------
// Pack weight matrix B[K x N] (f32, row stride Bs) into fragment-major bf16:
//   Bp[((kc*numTn + tn)*32 + lane)*16 + e],  e-th element of lane's fragment:
//   K = kc*32 + 16*(lane>=16) + e',  N = tn*16 + (lane&15)
// Zero-filled outside [K,N] so the GEMM needs no guards. numTn is padded to a
// multiple of 8 (one 128-col block).
// blockDim = 64: thread t -> lane = t&31, half = t>>5 (8 contiguous elements).
// ---------------------------------------------------------------------------
__global__ void k_packB(const float* __restrict__ B, int Bs, int N, int K,
                        int numTn, __bf16* __restrict__ Bp)
{
    const int tn   = blockIdx.x;
    const int kc   = blockIdx.y;
    const int lane = threadIdx.x & 31;
    const int half = threadIdx.x >> 5;
    const int hi   = lane >> 4;
    const int col  = tn * 16 + (lane & 15);
    const int kb   = kc * 32 + hi * 16 + half * 8;

    v8bf o;
    #pragma unroll
    for (int i = 0; i < 8; ++i) {
        int kk = kb + i;
        float v = (kk < K && col < N) ? B[(size_t)kk * Bs + col] : 0.0f;
        o[i] = (__bf16)v;
    }
    ((v8bf*)Bp)[((size_t)kc * numTn + tn) * 64 + (size_t)lane * 2 + half] = o;
}

// ---------------------------------------------------------------------------
// WMMA GEMM:  C[M x N] = f(A)[M x K] @ B[K x N] + bias
//   - M % 32 == 0 (100000, 4096 both qualify), K % 4 == 0 (300/600/1200)
//   - B pre-packed fragment-major bf16 (zero-padded), numTn % 8 == 0
//   - block: 128 threads = 4 waves; block tile 32(M) x 128(N)
//   - each wave: 2 M-tiles x 2 N-tiles = 4 accumulators, 4 wmma per k-step
// ---------------------------------------------------------------------------
template <int AMODE>
__global__ __launch_bounds__(128)
void k_gemm_wmma(const float* __restrict__ A,
                 const __bf16* __restrict__ Bp,
                 const float* __restrict__ bias,
                 float* __restrict__ C, int Cs,
                 int M, int N, int K, int numTn,
                 const float* __restrict__ scale,
                 const float* __restrict__ shift)
{
    __shared__ v16bf AsmV[64];           // 2 m-tiles x 32 lanes x 16 bf16 (2 KB)

    const int tid  = threadIdx.x;
    const int wave = tid >> 5;
    const int lane = tid & 31;

    const int m0  = blockIdx.x * 32;
    const int tn0 = blockIdx.y * 8 + wave * 2;     // first n-tile of this wave
    const int numKc = (K + 31) >> 5;

    // staging role: thread -> (m-tile, lane, half) -> 8 contiguous K elements
    const int mt_s    = tid >> 6;
    const int rem_s   = tid & 63;
    const int lane_s  = rem_s >> 1;
    const int half_s  = rem_s & 1;
    const int row_s   = m0 + mt_s * 16 + (lane_s & 15);
    const int kbase_s = (lane_s >> 4) * 16 + half_s * 8;
    const int sunit_s = (mt_s * 32 + lane_s) * 2 + half_s;

    // compute role
    const int hi    = lane >> 4;
    const int nlane = lane & 15;

    v8f acc00 = {}, acc01 = {}, acc10 = {}, acc11 = {};

    const v16bf* BpV = (const v16bf*)Bp;
    const float* arow = A + (size_t)row_s * K;

    for (int kc = 0; kc < numKc; ++kc) {
        const int kk0 = kc * 32 + kbase_s;

        // ---- stage A fragment slice: 8 contiguous f32 -> transform -> bf16 ----
        // K % 4 == 0 and kk0 % 8 == 0  =>  16-byte aligned float4 loads.
        // (Reads may run past K within the workspace; values are zero-selected
        //  below, and Bp is zero there anyway.)
        float4 f0 = *(const float4*)(arow + kk0);
        float4 f1 = *(const float4*)(arow + kk0 + 4);
        float vv[8] = {f0.x, f0.y, f0.z, f0.w, f1.x, f1.y, f1.z, f1.w};

        v8bf o;
        #pragma unroll
        for (int i = 0; i < 8; ++i) {
            int kk = kk0 + i;
            float x = (kk < K) ? vv[i] : 0.0f;          // NaN-safe pad
            if (AMODE == AMODE_BN_RELU) {
                int kcl = (kk < K) ? kk : (K - 1);
                x = fmaxf(x * scale[kcl] + shift[kcl], 0.0f);
            } else if (AMODE == AMODE_BN_GELU) {
                int kcl = (kk < K) ? kk : (K - 1);
                float t = x * scale[kcl] + shift[kcl];
                x = 0.5f * t * (1.0f + erff(t * 0.70710678118654752f));
            } else if (AMODE == AMODE_SILU) {
                x = x * __builtin_amdgcn_rcpf(1.0f + __expf(-x));
            }
            o[i] = (__bf16)x;
        }
        ((v8bf*)AsmV)[sunit_s] = o;                     // one ds_store_b128
        __syncthreads();

        // ---- fragments ----
        v16bf a0 = AsmV[lane];                          // m-tile 0
        v16bf a1 = AsmV[32 + lane];                     // m-tile 1
        size_t bb = ((size_t)kc * numTn + tn0) * 32 + lane;
        v16bf b0 = BpV[bb];
        v16bf b1 = BpV[bb + 32];

        acc00 = __builtin_amdgcn_wmma_f32_16x16x32_bf16(false, a0, false, b0,
                                                        (short)0, acc00, false, false);
        acc01 = __builtin_amdgcn_wmma_f32_16x16x32_bf16(false, a0, false, b1,
                                                        (short)0, acc01, false, false);
        acc10 = __builtin_amdgcn_wmma_f32_16x16x32_bf16(false, a1, false, b0,
                                                        (short)0, acc10, false, false);
        acc11 = __builtin_amdgcn_wmma_f32_16x16x32_bf16(false, a1, false, b1,
                                                        (short)0, acc11, false, false);
        __syncthreads();
    }

    // ---- epilogue: bias add + store (C 16x16 f32 layout) ----
    const int col0 = tn0 * 16 + nlane;
    const int col1 = col0 + 16;
    if (col0 < N) {
        float bv = bias ? bias[col0] : 0.0f;
        #pragma unroll
        for (int r = 0; r < 8; ++r) {
            int row = m0 + r + hi * 8;
            C[(size_t)row * Cs + col0]        = acc00[r] + bv;
            C[(size_t)(row + 16) * Cs + col0] = acc10[r] + bv;
        }
    }
    if (col1 < N) {
        float bv = bias ? bias[col1] : 0.0f;
        #pragma unroll
        for (int r = 0; r < 8; ++r) {
            int row = m0 + r + hi * 8;
            C[(size_t)row * Cs + col1]        = acc01[r] + bv;
            C[(size_t)(row + 16) * Cs + col1] = acc11[r] + bv;
        }
    }
}

// ---------------------------------------------------------------------------
// h = atom_emb[x_idx]
// ---------------------------------------------------------------------------
__global__ void k_gather(const float* __restrict__ atom_emb,
                         const int* __restrict__ x_idx,
                         float* __restrict__ h)
{
    size_t i = (size_t)blockIdx.x * blockDim.x + threadIdx.x;
    if (i >= (size_t)NN * DD) return;
    int n = (int)(i / DD);
    int d = (int)(i % DD);
    h[i] = atom_emb[(size_t)x_idx[n] * DD + d];
}

// ---------------------------------------------------------------------------
// edge scatter: agg[dst] += relu(h[src] + edge_emb[l][attr])
// ---------------------------------------------------------------------------
__global__ void k_scatter(const float* __restrict__ h,
                          const float* __restrict__ edge_emb, int l,
                          const int* __restrict__ src,
                          const int* __restrict__ dst,
                          const int* __restrict__ attr,
                          float* __restrict__ agg)
{
    int e = blockIdx.x;
    int s = src[e], t = dst[e], a = attr[e];
    const float* eb = edge_emb + ((size_t)l * 5 + a) * DD;
    const float* hs = h + (size_t)s * DD;
    float* ad = agg + (size_t)t * DD;
    for (int d = threadIdx.x; d < DD; d += blockDim.x) {
        float m = fmaxf(hs[d] + eb[d], 0.0f);
        atomicAdd(&ad[d], m);   // global_atomic_add_f32
    }
}

// ---------------------------------------------------------------------------
// z = (1+eps[l])*h + agg
// ---------------------------------------------------------------------------
__global__ void k_combine(const float* __restrict__ h,
                          const float* __restrict__ agg,
                          const float* __restrict__ eps, int l,
                          float* __restrict__ z)
{
    size_t i = (size_t)blockIdx.x * blockDim.x + threadIdx.x;
    if (i >= (size_t)NN * DD) return;
    z[i] = (1.0f + eps[l]) * h[i] + agg[i];
}

// ---------------------------------------------------------------------------
// column sums / sums-of-squares (coalesced: thread -> column)
// ---------------------------------------------------------------------------
__global__ void k_colstats(const float* __restrict__ X, int M, int N,
                           int rowsPer, float* __restrict__ sum,
                           float* __restrict__ sumsq)
{
    int j = blockIdx.x * blockDim.x + threadIdx.x;
    if (j >= N) return;
    int r0 = blockIdx.y * rowsPer;
    int r1 = r0 + rowsPer; if (r1 > M) r1 = M;
    float s = 0.0f, q = 0.0f;
    for (int r = r0; r < r1; ++r) {
        float v = X[(size_t)r * N + j];
        s += v; q += v * v;
    }
    atomicAdd(&sum[j], s);
    atomicAdd(&sumsq[j], q);
}

__global__ void k_stats_final(const float* __restrict__ sum,
                              const float* __restrict__ sumsq,
                              const float* __restrict__ g,
                              const float* __restrict__ b,
                              float* __restrict__ scale,
                              float* __restrict__ shift,
                              int N, float invM)
{
    int j = blockIdx.x * blockDim.x + threadIdx.x;
    if (j >= N) return;
    float mean = sum[j] * invM;
    float var  = sumsq[j] * invM - mean * mean;
    float s = g[j] * rsqrtf(var + BN_EPS);
    scale[j] = s;
    shift[j] = b[j] - mean * s;
}

// ---------------------------------------------------------------------------
// encoder tail: h += maybe_relu(bn(t2))
// ---------------------------------------------------------------------------
__global__ void k_bn_res(const float* __restrict__ t2,
                         const float* __restrict__ scale,
                         const float* __restrict__ shift,
                         int do_relu, float* __restrict__ h)
{
    size_t i = (size_t)blockIdx.x * blockDim.x + threadIdx.x;
    if (i >= (size_t)NN * DD) return;
    int d = (int)(i % DD);
    float v = t2[i] * scale[d] + shift[d];
    if (do_relu) v = fmaxf(v, 0.0f);
    h[i] += v;
}

// ---------------------------------------------------------------------------
// segment_max over sorted `batch` -> h_graph[G x D]
// ---------------------------------------------------------------------------
__device__ __forceinline__ int lower_bound_i(const int* b, int n, int v)
{
    int lo = 0, hi = n;
    while (lo < hi) { int m = (lo + hi) >> 1; if (b[m] < v) lo = m + 1; else hi = m; }
    return lo;
}

__global__ void k_segmax(const float* __restrict__ h,
                         const int* __restrict__ batch,
                         float* __restrict__ hg)
{
    int g = blockIdx.x;
    int start = lower_bound_i(batch, NN, g);
    int end   = lower_bound_i(batch, NN, g + 1);
    for (int d = threadIdx.x; d < DD; d += blockDim.x) {
        float m = -INFINITY;
        for (int r = start; r < end; ++r)
            m = fmaxf(m, h[(size_t)r * DD + d]);
        hg[(size_t)g * DD + d] = m;
    }
}

// ---------------------------------------------------------------------------
// split d -> mu (to d_out and ws) and sigma = softplus(d[:,D:]) + 1e-7
// ---------------------------------------------------------------------------
__global__ void k_mu_sigma(const float* __restrict__ dtmp,
                           float* __restrict__ out_mu,
                           float* __restrict__ out_sigma,
                           float* __restrict__ mu_ws)
{
    size_t i = (size_t)blockIdx.x * blockDim.x + threadIdx.x;
    if (i >= (size_t)GG * HH) return;
    int r = (int)(i / HH);
    int c = (int)(i % HH);
    float v = dtmp[i];
    if (c < DD) {
        out_mu[(size_t)r * DD + c] = v;
        mu_ws[(size_t)r * DD + c] = v;
    } else {
        float sp = (v > 0.0f) ? (v + log1pf(expf(-v))) : log1pf(expf(v));
        out_sigma[(size_t)r * DD + (c - DD)] = sp + 1e-7f;
    }
}

// ---------------------------------------------------------------------------
// host-side GEMM launcher: pack B, then dispatch templated WMMA GEMM
// ---------------------------------------------------------------------------
static inline void launch_gemm(const float* A, const float* W, int Ws,
                               const float* bias, float* C, int Cs,
                               int M, int N, int K, int amode,
                               const float* scale, const float* shift,
                               __bf16* Bp, hipStream_t s)
{
    int nBlkN  = (N + 127) / 128;
    int numTn  = nBlkN * 8;              // padded to a whole 128-col block
    int numKc  = (K + 31) / 32;
    k_packB<<<dim3(numTn, numKc), 64, 0, s>>>(W, Ws, N, K, numTn, Bp);

    dim3 grid(M / 32, nBlkN);
    switch (amode) {
    case AMODE_NONE:
        k_gemm_wmma<AMODE_NONE><<<grid, 128, 0, s>>>(A, Bp, bias, C, Cs, M, N, K, numTn, scale, shift); break;
    case AMODE_BN_RELU:
        k_gemm_wmma<AMODE_BN_RELU><<<grid, 128, 0, s>>>(A, Bp, bias, C, Cs, M, N, K, numTn, scale, shift); break;
    case AMODE_BN_GELU:
        k_gemm_wmma<AMODE_BN_GELU><<<grid, 128, 0, s>>>(A, Bp, bias, C, Cs, M, N, K, numTn, scale, shift); break;
    default:
        k_gemm_wmma<AMODE_SILU><<<grid, 128, 0, s>>>(A, Bp, bias, C, Cs, M, N, K, numTn, scale, shift); break;
    }
}

extern "C" void kernel_launch(void* const* d_in, const int* in_sizes, int n_in,
                              void* d_out, int out_size, void* d_ws, size_t ws_size,
                              hipStream_t stream)
{
    (void)in_sizes; (void)n_in; (void)out_size; (void)ws_size;

    const float* atom_emb   = (const float*)d_in[0];
    const float* edge_emb   = (const float*)d_in[1];
    const float* eps        = (const float*)d_in[2];
    const float* conv_w1    = (const float*)d_in[3];
    const float* conv_b1    = (const float*)d_in[4];
    const float* conv_bn1_g = (const float*)d_in[5];
    const float* conv_bn1_b = (const float*)d_in[6];
    const float* conv_w2    = (const float*)d_in[7];
    const float* conv_b2    = (const float*)d_in[8];
    const float* bn_g       = (const float*)d_in[9];
    const float* bn_b       = (const float*)d_in[10];
    const float* dist_w     = (const float*)d_in[11];
    const float* dist_b     = (const float*)d_in[12];
    const float* dec_w1     = (const float*)d_in[13];
    const float* dec_b1     = (const float*)d_in[14];
    const float* dec_bn_g   = (const float*)d_in[15];
    const float* dec_bn_b   = (const float*)d_in[16];
    const float* dec_w2     = (const float*)d_in[17];
    const float* dec_b2     = (const float*)d_in[18];
    const int*   x_idx      = (const int*)d_in[19];
    const int*   edge_attr  = (const int*)d_in[20];
    const int*   edge_index = (const int*)d_in[21];
    const int*   batch      = (const int*)d_in[22];

    const int* e_src = edge_index;
    const int* e_dst = edge_index + EE;

    float* out = (float*)d_out;

    // ---- workspace carve-up (floats) ----
    float* ws = (float*)d_ws;
    size_t off = 0;
    float* h    = ws + off; off += (size_t)NN * DD;   // 30,000,000
    float* zbuf = ws + off; off += (size_t)NN * DD;   // z, then reused for t2
    float* agg  = ws + off; off += (size_t)NN * DD;
    float* t1   = ws + off; off += (size_t)NN * HH;   // 60,000,000
    float* hg   = ws + off; off += (size_t)GG * DD;
    float* dtmp = ws + off; off += (size_t)GG * HH;
    float* muws = ws + off; off += (size_t)GG * DD;
    float* t3   = ws + off; off += (size_t)GG * DHH;
    __bf16* Bp  = (__bf16*)(ws + off); off += 1200000; // packed B (<= 4.36 MB)
    float* ssum = ws + off; off += DHH;
    float* ssq  = ws + off; off += DHH;
    float* bscale = ws + off; off += DHH;
    float* bshift = ws + off; off += DHH;
    off += 64;                                        // tail slack for f32x4 over-reads

    const int TPB = 256;
    const size_t ND = (size_t)NN * DD;

    // ---- h = atom_emb[x_idx] ----
    k_gather<<<(unsigned)((ND + TPB - 1) / TPB), TPB, 0, stream>>>(atom_emb, x_idx, h);

    // ---- encoder: 5 GIN layers ----
    for (int l = 0; l < LL; ++l) {
        hipMemsetAsync(agg, 0, ND * sizeof(float), stream);
        k_scatter<<<EE, TPB, 0, stream>>>(h, edge_emb, l, e_src, e_dst, edge_attr, agg);
        k_combine<<<(unsigned)((ND + TPB - 1) / TPB), TPB, 0, stream>>>(h, agg, eps, l, zbuf);

        // t1 = z @ w1 + b1     (N x 300) @ (300 x 600)
        launch_gemm(zbuf, conv_w1 + (size_t)l * DD * HH, HH,
                    conv_b1 + (size_t)l * HH, t1, HH,
                    NN, HH, DD, AMODE_NONE, nullptr, nullptr, Bp, stream);

        // BN1 stats over N rows of t1
        hipMemsetAsync(ssum, 0, 2 * DHH * sizeof(float), stream);
        {
            int rowsPer = (NN + 127) / 128;
            dim3 g((HH + TPB - 1) / TPB, 128);
            k_colstats<<<g, TPB, 0, stream>>>(t1, NN, HH, rowsPer, ssum, ssq);
        }
        k_stats_final<<<(HH + TPB - 1) / TPB, TPB, 0, stream>>>(
            ssum, ssq, conv_bn1_g + (size_t)l * HH, conv_bn1_b + (size_t)l * HH,
            bscale, bshift, HH, 1.0f / NN);

        // t2 = relu(bn(t1)) @ w2 + b2   (N x 600) @ (600 x 300), BN+ReLU fused on A
        launch_gemm(t1, conv_w2 + (size_t)l * HH * DD, DD,
                    conv_b2 + (size_t)l * DD, zbuf, DD,
                    NN, DD, HH, AMODE_BN_RELU, bscale, bshift, Bp, stream);

        // BN2 stats over N rows of t2
        hipMemsetAsync(ssum, 0, 2 * DHH * sizeof(float), stream);
        {
            int rowsPer = (NN + 127) / 128;
            dim3 g((DD + TPB - 1) / TPB, 128);
            k_colstats<<<g, TPB, 0, stream>>>(zbuf, NN, DD, rowsPer, ssum, ssq);
        }
        k_stats_final<<<(DD + TPB - 1) / TPB, TPB, 0, stream>>>(
            ssum, ssq, bn_g + (size_t)l * DD, bn_b + (size_t)l * DD,
            bscale, bshift, DD, 1.0f / NN);

        // h += maybe_relu(bn(t2))
        k_bn_res<<<(unsigned)((ND + TPB - 1) / TPB), TPB, 0, stream>>>(
            zbuf, bscale, bshift, (l < LL - 1) ? 1 : 0, h);
    }

    // ---- graph pooling ----
    k_segmax<<<GG, 128, 0, stream>>>(h, batch, hg);

    // ---- distribution head: dtmp = silu(hg) @ dist_w + dist_b ----
    launch_gemm(hg, dist_w, HH, dist_b, dtmp, HH,
                GG, HH, DD, AMODE_SILU, nullptr, nullptr, Bp, stream);

    float* out_mu    = out;                       // G x 300
    float* out_sigma = out + (size_t)GG * DD;     // G x 300
    k_mu_sigma<<<(unsigned)(((size_t)GG * HH + TPB - 1) / TPB), TPB, 0, stream>>>(
        dtmp, out_mu, out_sigma, muws);

    // ---- decoders ----
    const int   dec_dims[6] = {1024, 1111, 862, 1783, 966, 978};
    float* dec_base[6];
    int    dec_stride[6];
    {
        size_t b2 = (size_t)GG * 600;                 // outs[0]
        size_t bg = b2 + (size_t)GG * 1024;           // out_gene (stride 1973)
        size_t bc = bg + (size_t)GG * 1973;           // out_cell (stride 2749)
        size_t b5 = bc + (size_t)GG * 2749;           // outs[5]
        dec_base[0] = out + b2;         dec_stride[0] = 1024;
        dec_base[1] = out + bg;         dec_stride[1] = 1973;
        dec_base[2] = out + bg + 1111;  dec_stride[2] = 1973;
        dec_base[3] = out + bc;         dec_stride[3] = 2749;
        dec_base[4] = out + bc + 1783;  dec_stride[4] = 2749;
        dec_base[5] = out + b5;         dec_stride[5] = 978;
    }

    for (int i = 0; i < 6; ++i) {
        // t3 = mu @ dec_w1[i] + dec_b1[i]   (4096 x 300) @ (300 x 1200)
        launch_gemm(muws, dec_w1 + (size_t)i * DD * DHH, DHH,
                    dec_b1 + (size_t)i * DHH, t3, DHH,
                    GG, DHH, DD, AMODE_NONE, nullptr, nullptr, Bp, stream);

        // BN stats over G rows
        hipMemsetAsync(ssum, 0, 2 * DHH * sizeof(float), stream);
        {
            int rowsPer = (GG + 15) / 16;
            dim3 g((DHH + TPB - 1) / TPB, 16);
            k_colstats<<<g, TPB, 0, stream>>>(t3, GG, DHH, rowsPer, ssum, ssq);
        }
        k_stats_final<<<(DHH + TPB - 1) / TPB, TPB, 0, stream>>>(
            ssum, ssq, dec_bn_g + (size_t)i * DHH, dec_bn_b + (size_t)i * DHH,
            bscale, bshift, DHH, 1.0f / GG);

        // out = gelu(bn(t3)) @ dec_w2[i][:, :dims] + dec_b2[i][:dims]
        launch_gemm(t3, dec_w2 + (size_t)i * DHH * MAXO, MAXO,
                    dec_b2 + (size_t)i * MAXO,
                    dec_base[i], dec_stride[i],
                    GG, dec_dims[i], DHH, AMODE_BN_GELU, bscale, bshift, Bp, stream);
    }
}
